// AttnEmbedder_42245298323749
// MI455X (gfx1250) — matrix operand order repaired and checked by reading, not compiled, not run on hardware
//
#include <hip/hip_runtime.h>

typedef __attribute__((ext_vector_type(16))) _Float16 v16h;
typedef __attribute__((ext_vector_type(8)))  _Float16 v8h;
typedef __attribute__((ext_vector_type(8)))  float    v8f;

#define N_PARAMS 130048
#define EPS 1e-5f

// ---- dynamic LDS layout (bytes), all 32B-aligned sections ----
#define SKV_OFF   0                      // 160*128 f32 = 81920  (K cols 0..63, V cols 64..127)
#define SQ0_OFF   (81920)                // 64  f32
#define SBIAS_OFF (SQ0_OFF + 256)        // 192 f32
#define SSCAL_OFF (SBIAS_OFF + 768)      // 160 f32
#define SATTN_OFF (SSCAL_OFF + 640)      // 64  f32
#define SH_OFF    (SATTN_OFF + 256)      // 64  f32
#define SF_OFF    (SH_OFF + 256)         // 96  f32
#define ST_OFF    (SF_OFF + 384)         // 160*64 f16 = 20480   (token matrix, padded)
#define SW_OFF    (ST_OFF + 20480)       // 192*64 f16 = 24576   (in_proj_w as-is == B^T)
#define LDS_BYTES (SW_OFF + 24576)       // 129536 B  (<320KB/WGP -> 2 WG/WGP)

__global__ __launch_bounds__(128) void attn_embedder_kernel(
    const float* __restrict__ x, const float* __restrict__ pos_emb,
    const float* __restrict__ comp_token,
    const float* __restrict__ in_proj_w, const float* __restrict__ in_proj_b,
    const float* __restrict__ out_proj_w, const float* __restrict__ out_proj_b,
    const float* __restrict__ lin1_w, const float* __restrict__ lin1_b,
    const float* __restrict__ lin2_w, const float* __restrict__ lin2_b,
    const float* __restrict__ ln1_w, const float* __restrict__ ln1_b,
    const float* __restrict__ ln2_w, const float* __restrict__ ln2_b,
    float* __restrict__ out)
{
  extern __shared__ __align__(32) char smem[];
  float*    sKV   = (float*)(smem + SKV_OFF);
  float*    sQ0   = (float*)(smem + SQ0_OFF);
  float*    sBias = (float*)(smem + SBIAS_OFF);
  float*    sScal = (float*)(smem + SSCAL_OFF);
  float*    sAttn = (float*)(smem + SATTN_OFF);
  float*    sH    = (float*)(smem + SH_OFF);
  float*    sF    = (float*)(smem + SF_OFF);
  _Float16* sT    = (_Float16*)(smem + ST_OFF);
  _Float16* sW    = (_Float16*)(smem + SW_OFF);

  const int tid  = threadIdx.x;
  const int lane = tid & 31;
  const int wave = tid >> 5;

  // ---- map block -> (layer, batch, kernel) ----
  int sid = blockIdx.x;
  int start, L, kno, bb, k, koff;
  if (sid < 1024)      { start = 0;     L = 72;  kno = 256; bb = sid >> 8; k = sid & 255; koff = 0;   }
  else if (sid < 2048) { int s = sid - 1024; start = 18688; L = 144; kno = 256; bb = s >> 8; k = s & 255; koff = 256; }
  else                 { int s = sid - 2048; start = 55808; L = 144; kno = 512; bb = s >> 9; k = s & 511; koff = 512; }
  const int n     = L + 1;            // scalars per sequence (window + bias)
  const int S     = n + 1;            // tokens incl. CLS
  const int S_pad = (S + 15) & ~15;   // 80 or 160

  // ---- stage weights (f16), bias, gathered scalars ----
  for (int i = tid; i < 192 * 64; i += 128) sW[i] = (_Float16)in_proj_w[i];
  for (int i = tid; i < 192; i += 128)      sBias[i] = in_proj_b[i];
  const float* xb = x + bb * N_PARAMS;
  for (int i = tid; i < n; i += 128)
    sScal[i] = (i < L) ? xb[start + k * L + i] : xb[start + kno * L + k];
  __syncthreads();

  // ---- build f16 token matrix: row 0 = CLS, rows 1..n = scalar + pos_emb, pad = 0 ----
  for (int i = tid; i < S_pad * 64; i += 128) {
    int r = i >> 6, d = i & 63;
    float v;
    if (r == 0)      v = comp_token[d];
    else if (r <= n) v = sScal[r - 1] + pos_emb[(r - 1) * 64 + d];
    else             v = 0.0f;
    sT[i] = (_Float16)v;
  }
  __syncthreads();

  // ---- QKV projection via v_wmma_f32_16x16x32_f16 (K=64 -> 2 chained WMMAs/tile) ----
  // Only needed tiles: n-tiles 4..11 (K,V) for all m-tiles; n-tiles 0..3 (Q) at m-tile 0.
  {
    const int Smt   = S_pad >> 4;
    const int total = 4 + Smt * 8;
    for (int idx = wave; idx < total; idx += 4) {
      int mt, nt;
      if (idx < 4) { mt = 0; nt = idx; }
      else { int j = idx - 4; mt = j >> 3; nt = 4 + (j & 7); }
      const int m  = mt * 16 + (lane & 15);
      const int nn = nt * 16 + (lane & 15);
      const int kh = lane >> 4;
      // A fragment (16x32 f16, contiguous K to match B): lanes 0-15 K{0..15}, lanes 16-31 K{16..31}
      union { v16h v; v8h h[2]; } a0, a1;
      const _Float16* rp = sT + m * 64;
      a0.h[0] = *(const v8h*)(rp + kh * 16);
      a0.h[1] = *(const v8h*)(rp + 8 + kh * 16);
      a1.h[0] = *(const v8h*)(rp + 32 + kh * 16);
      a1.h[1] = *(const v8h*)(rp + 40 + kh * 16);
      // B fragment (32x16): column nn, contiguous K run in sW (== in_proj_w row nn)
      const v16h b0 = *(const v16h*)(sW + nn * 64 + kh * 16);
      const v16h b1 = *(const v16h*)(sW + nn * 64 + 32 + kh * 16);
      v8f c = {};
      c = __builtin_amdgcn_wmma_f32_16x16x32_f16(false, a0.v, false, b0, (short)0, c, false, false);
      c = __builtin_amdgcn_wmma_f32_16x16x32_f16(false, a1.v, false, b1, (short)0, c, false, false);
      const float bias = sBias[nn];
      if (nt >= 4) {                         // K/V tiles -> full store
        const int col = nt * 16 - 64 + (lane & 15);
        #pragma unroll
        for (int r = 0; r < 8; ++r) {
          const int mr = mt * 16 + r + 8 * kh;  // D layout: VGPR r -> row r (+8 upper half)
          sKV[mr * 128 + col] = c[r] + bias;
        }
      } else if (lane < 16) {                // Q tile -> only CLS row (m==0: r==0, lanes 0-15)
        sQ0[nn] = c[0] + bias;
      }
    }
  }
  __syncthreads();

  // ---- CLS-row attention: one head per wave, lanes stripe key rows ----
  {
    const int h = wave;
    float q[16];
    #pragma unroll
    for (int d = 0; d < 16; ++d) q[d] = sQ0[h * 16 + d];
    float sc[5];
    int cnt = 0;
    float mx = -3.0e38f;
    for (int t = lane; t < S; t += 32) {
      const float* kr = sKV + t * 128 + h * 16;
      float s = 0.f;
      #pragma unroll
      for (int d = 0; d < 16; ++d) s += q[d] * kr[d];
      s *= 0.25f;                                  // 1/sqrt(D_HEAD)
      sc[cnt++] = s;
      mx = fmaxf(mx, s);
    }
    #pragma unroll
    for (int off = 16; off > 0; off >>= 1) mx = fmaxf(mx, __shfl_xor(mx, off, 32));
    float sum = 0.f, acc[16];
    #pragma unroll
    for (int d = 0; d < 16; ++d) acc[d] = 0.f;
    int ci = 0;
    for (int t = lane; t < S; t += 32) {
      float p = __expf(sc[ci++] - mx);
      sum += p;
      const float* vr = sKV + t * 128 + 64 + h * 16;
      #pragma unroll
      for (int d = 0; d < 16; ++d) acc[d] += p * vr[d];
    }
    #pragma unroll
    for (int off = 16; off > 0; off >>= 1) {
      sum += __shfl_xor(sum, off, 32);
      #pragma unroll
      for (int d = 0; d < 16; ++d) acc[d] += __shfl_xor(acc[d], off, 32);
    }
    if (lane == 0) {
      const float inv = 1.0f / sum;
      #pragma unroll
      for (int d = 0; d < 16; ++d) sAttn[h * 16 + d] = acc[d] * inv;
    }
  }
  __syncthreads();

  // ---- CLS-row tail on wave 0: out_proj, LN1, FFN, LN2, tiled write ----
  if (wave == 0) {
    const int e0 = lane, e1 = lane + 32;
    float o0 = out_proj_b[e0], o1 = out_proj_b[e1];
    for (int c = 0; c < 64; ++c) {
      const float a = sAttn[c];
      o0 += a * out_proj_w[e0 * 64 + c];
      o1 += a * out_proj_w[e1 * 64 + c];
    }
    float p0 = comp_token[e0] + o0;
    float p1 = comp_token[e1] + o1;
    float s = p0 + p1;
    #pragma unroll
    for (int off = 16; off > 0; off >>= 1) s += __shfl_xor(s, off, 32);
    const float mu = s * (1.0f / 64.0f);
    float dv = (p0 - mu) * (p0 - mu) + (p1 - mu) * (p1 - mu);
    #pragma unroll
    for (int off = 16; off > 0; off >>= 1) dv += __shfl_xor(dv, off, 32);
    const float rs = rsqrtf(dv * (1.0f / 64.0f) + EPS);
    const float h0 = (p0 - mu) * rs * ln1_w[e0] + ln1_b[e0];
    const float h1 = (p1 - mu) * rs * ln1_w[e1] + ln1_b[e1];
    sH[e0] = h0; sH[e1] = h1;   // same-wave LDS ops are in-order
    #pragma unroll
    for (int j = 0; j < 3; ++j) {
      const int f = lane + j * 32;
      float v = lin1_b[f];
      for (int c = 0; c < 64; ++c) v += sH[c] * lin1_w[f * 64 + c];
      sF[f] = fmaxf(v, 0.0f);
    }
    float f20 = lin2_b[e0], f21 = lin2_b[e1];
    for (int f = 0; f < 96; ++f) {
      const float vf = sF[f];
      f20 += vf * lin2_w[e0 * 96 + f];
      f21 += vf * lin2_w[e1 * 96 + f];
    }
    const float q0 = h0 + f20, q1 = h1 + f21;
    float s2 = q0 + q1;
    #pragma unroll
    for (int off = 16; off > 0; off >>= 1) s2 += __shfl_xor(s2, off, 32);
    const float mu2 = s2 * (1.0f / 64.0f);
    float dv2 = (q0 - mu2) * (q0 - mu2) + (q1 - mu2) * (q1 - mu2);
    #pragma unroll
    for (int off = 16; off > 0; off >>= 1) dv2 += __shfl_xor(dv2, off, 32);
    const float rs2 = rsqrtf(dv2 * (1.0f / 64.0f) + EPS);
    const float y0 = (q0 - mu2) * rs2 * ln2_w[e0] + ln2_b[e0];
    const float y1 = (q1 - mu2) * rs2 * ln2_w[e1] + ln2_b[e1];
    float* ob = out + ((size_t)bb * 1024 + (size_t)(koff + k)) * 128;
    ob[lane]      = y0;
    ob[lane + 32] = y1;
    ob[lane + 64] = y0;   // tile(1,1,2): out[..., c] = y[c % 64]
    ob[lane + 96] = y1;
  }
}

extern "C" void kernel_launch(void* const* d_in, const int* in_sizes, int n_in,
                              void* d_out, int out_size, void* d_ws, size_t ws_size,
                              hipStream_t stream) {
  const float* x          = (const float*)d_in[0];
  const float* pos_emb    = (const float*)d_in[1];
  const float* comp_token = (const float*)d_in[2];
  const float* in_proj_w  = (const float*)d_in[3];
  const float* in_proj_b  = (const float*)d_in[4];
  const float* out_proj_w = (const float*)d_in[5];
  const float* out_proj_b = (const float*)d_in[6];
  const float* lin1_w     = (const float*)d_in[7];
  const float* lin1_b     = (const float*)d_in[8];
  const float* lin2_w     = (const float*)d_in[9];
  const float* lin2_b     = (const float*)d_in[10];
  const float* ln1_w      = (const float*)d_in[11];
  const float* ln1_b      = (const float*)d_in[12];
  const float* ln2_w      = (const float*)d_in[13];
  const float* ln2_b      = (const float*)d_in[14];
  float* out = (float*)d_out;

  attn_embedder_kernel<<<dim3(4096), dim3(128), LDS_BYTES, stream>>>(
      x, pos_emb, comp_token, in_proj_w, in_proj_b, out_proj_w, out_proj_b,
      lin1_w, lin1_b, lin2_w, lin2_b, ln1_w, ln1_b, ln2_w, ln2_b, out);
}
